// TransNet_6725918785892
// MI455X (gfx1250) — matrix operand order repaired and checked by reading, not compiled, not run on hardware
//
#include <hip/hip_runtime.h>
#include <hip/hip_bf16.h>
#include <math.h>

// ---------------------------------------------------------------------------
// TransNet forward for MI455X (gfx1250, wave32, WMMA).
//
// Conv = implicit GEMM on v_wmma_f32_16x16x32_f16 with K ordered channel-
// fastest (k = tap*CinPad + c). Activations are NHWC f16 (channels padded to
// a multiple of 16): each lane's 16-K B-fragment is one 32B contiguous vector
// load and the 8-row accumulator store is one 16B vector store. Weights are
// repacked once per call into exact A-fragment order (f16). All layer shapes
// are template parameters (no runtime division). Each wave computes MBLK
// (<=4) M-tiles over one shared B fragment to cut activation re-reads 4x.
// ---------------------------------------------------------------------------

#define BATCH 2
#define HH 256
#define WW 256
#define NPIX (HH * WW)
#define NUM_SEG 1024
#define S1 (NUM_SEG + 1)

typedef __attribute__((ext_vector_type(16))) _Float16 v16h;
typedef __attribute__((ext_vector_type(8)))  _Float16 v8h;
typedef __attribute__((ext_vector_type(8)))  float    v8f;

// ---------------------------------------------------------------------------
// Weight repack: fp32 [CoutTot, CinT, Kf, Kf] -> f16 A-fragments.
// Fragment order: idx = ((chunk*TilesM + tile)*32 + lane)*16 + j
// GEMM K order:   k = tap*CinPad + c   (tap = ky*Kf + kx, channel-fastest)
// A 16x32 layout (matches B): lane&15 = M row; element j -> K j;
// lanes>=16 add K offset 16.
// ---------------------------------------------------------------------------
__global__ void pack_w_kernel(const float* __restrict__ w, _Float16* __restrict__ pk,
                              int CoutTot, int CinT, int CinPad, int KK, int nch) {
  int idx = blockIdx.x * blockDim.x + threadIdx.x;
  int tilesM = CoutTot >> 4;
  int total = nch * tilesM * 512;
  if (idx >= total) return;
  int j     = idx & 15;
  int lane  = (idx >> 4) & 31;
  int tl    = (idx >> 9) % tilesM;
  int chunk = (idx >> 9) / tilesM;
  int m  = tl * 16 + (lane & 15);
  int kl = j + ((lane & 16) ? 16 : 0);
  int k  = chunk * 32 + kl;
  int t  = k / CinPad;
  int c  = k - t * CinPad;
  float v = 0.0f;
  if (t < KK && c < CinT) v = w[((size_t)m * CinT + c) * KK + t];
  pk[idx] = (_Float16)v;
}

// ---------------------------------------------------------------------------
// Gated conv, implicit GEMM. One wave32 computes MBLK 16(M)x16(N) tiles that
// share one B (activation) fragment per K-chunk. Virtual input channel c:
// c < CinA -> srcA (spatial >>UPA for 2x nearest up), else skip srcB.
// ---------------------------------------------------------------------------
template <int HOUT, int WOUT, int CH, int CinA, int CinB, int CinPad,
          int KF, int STRIDE, int UPA, bool GATED, int MBLK>
__global__ __launch_bounds__(32) void gconv_wmma(
    const _Float16* __restrict__ srcA, const _Float16* __restrict__ srcB,
    const _Float16* __restrict__ wpk, const float* __restrict__ bias,
    _Float16* __restrict__ out)
{
  constexpr int KK   = KF * KF;
  constexpr int Kdim = KK * CinPad;
  constexpr int NCH  = (Kdim + 31) / 32;
  constexpr int PAD  = KF / 2;
  constexpr int HIN  = HOUT * STRIDE;
  constexpr int WIN  = WOUT * STRIDE;
  constexpr int CoutTot = GATED ? 2 * CH : CH;
  constexpr int TILESM  = CoutTot / 16;
  constexpr int CHT     = CH / 16;
  constexpr int HA = HIN >> UPA;
  constexpr int WA = WIN >> UPA;

  const int lane   = threadIdx.x;
  const int tileN  = blockIdx.x;
  const int tileM0 = blockIdx.y * MBLK;   // first of MBLK M-tiles for this wave

  // 16 pixels of this tile are consecutive in x on one row (W % 16 == 0).
  const int n   = tileN * 16 + (lane & 15);
  const int bb  = n / (HOUT * WOUT);
  const int rem = n % (HOUT * WOUT);
  const int oy  = rem / WOUT;
  const int ox  = rem % WOUT;

  const int hiB = (lane & 16) ? 16 : 0;  // contiguous K (same as A pack): upper lanes get K+16

  v8f accF[MBLK];
  v8f accG[MBLK];
  #pragma unroll
  for (int mb = 0; mb < MBLK; ++mb) { accF[mb] = (v8f){}; accG[mb] = (v8f){}; }

  for (int k0 = 0; k0 < NCH; ++k0) {
    // ---- B fragment: 16 consecutive K = one tap, 16 consecutive channels ----
    v16h bm = {};
    const int k16 = k0 * 32 + hiB;
    if (k16 < Kdim) {
      const int t  = k16 / CinPad;          // compile-time divisor
      const int c0 = k16 - t * CinPad;
      const int ky = t / KF;
      const int kx = t - ky * KF;
      const int iy = oy * STRIDE + ky - PAD;
      const int ix = ox * STRIDE + kx - PAD;
      if (iy >= 0 && iy < HIN && ix >= 0 && ix < WIN) {
        if (CinB == 0 || c0 < CinA) {
          bm = *(const v16h*)(srcA +
              (((size_t)bb * HA + (iy >> UPA)) * WA + (ix >> UPA)) * CinA + c0);
        } else {
          bm = *(const v16h*)(srcB +
              (((size_t)bb * HIN + iy) * WIN + ix) * CinB + (c0 - CinA));
        }
      }
    }

    // ---- A fragments (pre-packed, one vector load each); MBLK M-tiles
    //      amortize the B gather across up to 4x more WMMA issue. ----
    #pragma unroll
    for (int mb = 0; mb < MBLK; ++mb) {
      const v16h af = *(const v16h*)(wpk +
          (((size_t)k0 * TILESM + (tileM0 + mb)) * 32 + lane) * 16);
      accF[mb] = __builtin_amdgcn_wmma_f32_16x16x32_f16(false, af, false, bm,
                                                        (short)0, accF[mb],
                                                        false, false);
      if (GATED) {
        const v16h ag = *(const v16h*)(wpk +
            (((size_t)k0 * TILESM + (CHT + tileM0 + mb)) * 32 + lane) * 16);
        accG[mb] = __builtin_amdgcn_wmma_f32_16x16x32_f16(false, ag, false, bm,
                                                          (short)0, accG[mb],
                                                          false, false);
      }
    }
  }

  // ---- epilogue: C/D f32 16x16 (lane&15 = N col, vgpr v + 8*(lane>=16) = M row)
  const int mh = (lane & 16) ? 8 : 0;
  const size_t pixBase = (((size_t)bb * HOUT + oy) * WOUT + ox) * CH;
  #pragma unroll
  for (int mb = 0; mb < MBLK; ++mb) {
    v8h ov;
    #pragma unroll
    for (int v = 0; v < 8; ++v) {
      const int m = (tileM0 + mb) * 16 + mh + v;
      float o;
      if (GATED) {
        float f  = accF[mb][v] + bias[m];
        float g  = accG[mb][v] + bias[CH + m];
        float fr = f > 0.0f ? f : 0.0f;
        o = fr * (1.0f / (1.0f + __expf(-g)));
      } else {
        o = accF[mb][v] + bias[m];
      }
      ov[v] = (_Float16)o;
    }
    *(v8h*)(out + pixBase + (tileM0 + mb) * 16 + mh) = ov;
  }
}

// ---------------------------------------------------------------------------
// Glue kernels
// ---------------------------------------------------------------------------
__global__ void zero_kernel(float* p, int n) {
  int i = blockIdx.x * blockDim.x + threadIdx.x;
  if (i < n) p[i] = 0.0f;
}

__global__ void prep_seg_kernel(const float* __restrict__ tar,
                                const int* __restrict__ seg_in,
                                int* __restrict__ seg, int total) {
  int i = blockIdx.x * blockDim.x + threadIdx.x;
  if (i >= total) return;
  int b = i / NPIX, n = i - b * NPIX;
  float t0 = tar[((size_t)b * 2 + 0) * NPIX + n];
  float t1 = tar[((size_t)b * 2 + 1) * NPIX + n];
  int s = seg_in[i];
  if (t0 < 0.0f || t1 < 0.0f || s < 0) s = NUM_SEG;
  seg[i] = s;
}

__global__ void seg_count_kernel(const int* __restrict__ seg,
                                 float* __restrict__ counts, int total) {
  int i = blockIdx.x * blockDim.x + threadIdx.x;
  if (i >= total) return;
  int b = i / NPIX;
  atomicAdd(&counts[b * S1 + seg[i]], 1.0f);
}

// planar fp32 [B,C,N] scatter-add
__global__ void seg_sum_kernel(const float* __restrict__ x,
                               const int* __restrict__ seg,
                               float* __restrict__ sums, int C, int total) {
  int i = blockIdx.x * blockDim.x + threadIdx.x;
  if (i >= total) return;
  int b = i / NPIX, n = i - b * NPIX;
  int s = seg[i];
  for (int c = 0; c < C; ++c)
    atomicAdd(&sums[((size_t)b * C + c) * S1 + s],
              x[((size_t)b * C + c) * NPIX + n]);
}

// NHWC f16, C=16 scatter-add (UNet feature maps)
__global__ void seg_sum_feat_kernel(const _Float16* __restrict__ feat,
                                    const int* __restrict__ seg,
                                    float* __restrict__ sums, int total) {
  int i = blockIdx.x * blockDim.x + threadIdx.x;
  if (i >= total) return;
  int b = i / NPIX;
  int s = seg[i];
  const _Float16* f = feat + (size_t)i * 16;
  #pragma unroll
  for (int c = 0; c < 16; ++c)
    atomicAdd(&sums[((size_t)b * 16 + c) * S1 + s], (float)f[c]);
}

__global__ void seg_mean_kernel(float* __restrict__ sums,
                                const float* __restrict__ counts, int C) {
  int i = blockIdx.x * blockDim.x + threadIdx.x;
  int total = BATCH * C * S1;
  if (i >= total) return;
  int b = i / (C * S1);
  int s = i % S1;
  float cnt = counts[b * S1 + s];
  sums[i] = sums[i] / fmaxf(cnt, 1.0f);
}

// pos0 = where(valid, [src - c_src, tar - c_tar], -1) ++ mask, NHWC f16 C=16.
__global__ void build_in0_kernel(const float* __restrict__ src,
                                 const float* __restrict__ tar,
                                 const float* __restrict__ mask,
                                 const int* __restrict__ seg,
                                 const float* __restrict__ meanS,
                                 const float* __restrict__ meanT,
                                 _Float16* __restrict__ in0, int total) {
  int i = blockIdx.x * blockDim.x + threadIdx.x;
  if (i >= total) return;
  int b = i / NPIX, n = i - b * NPIX;
  int s = seg[i];
  float t0 = tar[((size_t)b * 2 + 0) * NPIX + n];
  float t1 = tar[((size_t)b * 2 + 1) * NPIX + n];
  bool valid = (t0 >= 0.0f) && (t1 >= 0.0f);
  float s0 = src[((size_t)b * 2 + 0) * NPIX + n];
  float s1 = src[((size_t)b * 2 + 1) * NPIX + n];
  float cs0 = meanS[((size_t)b * 2 + 0) * S1 + s];
  float cs1 = meanS[((size_t)b * 2 + 1) * S1 + s];
  float ct0 = meanT[((size_t)b * 2 + 0) * S1 + s];
  float ct1 = meanT[((size_t)b * 2 + 1) * S1 + s];
  _Float16* o = in0 + (size_t)i * 16;
  o[0] = (_Float16)(valid ? (s0 - cs0) : -1.0f);
  o[1] = (_Float16)(valid ? (s1 - cs1) : -1.0f);
  o[2] = (_Float16)(valid ? (t0 - ct0) : -1.0f);
  o[3] = (_Float16)(valid ? (t1 - ct1) : -1.0f);
  o[4] = (_Float16)mask[(size_t)b * NPIX + n];
  #pragma unroll
  for (int c = 5; c < 16; ++c) o[c] = (_Float16)0.0f;
}

// mreg: x[B,16,S1] fp32 -> relu(W1 x + b1) -> W2 h + b2 -> out[B,2,S1]
__global__ void mreg_kernel(const float* __restrict__ x,
                            const float* __restrict__ w1, const float* __restrict__ b1,
                            const float* __restrict__ w2, const float* __restrict__ b2,
                            float* __restrict__ out) {
  int i = blockIdx.x * blockDim.x + threadIdx.x;
  if (i >= BATCH * S1) return;
  int b = i / S1, s = i - b * S1;
  float xin[16];
  #pragma unroll
  for (int c = 0; c < 16; ++c) xin[c] = x[((size_t)b * 16 + c) * S1 + s];
  float h[64];
  for (int o = 0; o < 64; ++o) {
    float acc = b1[o];
    #pragma unroll
    for (int c = 0; c < 16; ++c) acc += w1[o * 16 + c] * xin[c];
    h[o] = acc > 0.0f ? acc : 0.0f;
  }
  for (int o = 0; o < 2; ++o) {
    float acc = b2[o];
    for (int c = 0; c < 64; ++c) acc += w2[o * 64 + c] * h[c];
    out[((size_t)b * 2 + o) * S1 + s] = acc;
  }
}

// R_mat = stack([r, r@R90], -1) + I = [[1+r0,-r1],[r1,1+r0]] -> [B,1024,2,2]
__global__ void write_R_kernel(const float* __restrict__ pred, float* __restrict__ outR) {
  int i = blockIdx.x * blockDim.x + threadIdx.x;
  if (i >= BATCH * NUM_SEG) return;
  int b = i / NUM_SEG, s = i - b * NUM_SEG;
  float r0 = pred[((size_t)b * 2 + 0) * S1 + s];
  float r1 = pred[((size_t)b * 2 + 1) * S1 + s];
  float* o = outR + (size_t)i * 4;
  o[0] = 1.0f + r0; o[1] = -r1; o[2] = r1; o[3] = 1.0f + r0;
}

// pos1 = where(valid, [src @ R(seg), tar], -1) ++ mask, NHWC f16 C=16.
__global__ void build_pos1_kernel(const float* __restrict__ src,
                                  const float* __restrict__ tar,
                                  const float* __restrict__ mask,
                                  const int* __restrict__ seg,
                                  const float* __restrict__ pred,
                                  _Float16* __restrict__ in1, int total) {
  int i = blockIdx.x * blockDim.x + threadIdx.x;
  if (i >= total) return;
  int b = i / NPIX, n = i - b * NPIX;
  int s = seg[i];
  float r0 = pred[((size_t)b * 2 + 0) * S1 + s];
  float r1 = pred[((size_t)b * 2 + 1) * S1 + s];
  float t0 = tar[((size_t)b * 2 + 0) * NPIX + n];
  float t1 = tar[((size_t)b * 2 + 1) * NPIX + n];
  bool valid = (t0 >= 0.0f) && (t1 >= 0.0f);
  float s0 = src[((size_t)b * 2 + 0) * NPIX + n];
  float s1 = src[((size_t)b * 2 + 1) * NPIX + n];
  // out_d = sum_c pos_c * R[c,d],  R = [[1+r0,-r1],[r1,1+r0]]
  float rot0 = s0 * (1.0f + r0) + s1 * r1;
  float rot1 = s0 * (-r1) + s1 * (1.0f + r0);
  _Float16* o = in1 + (size_t)i * 16;
  o[0] = (_Float16)(valid ? rot0 : -1.0f);
  o[1] = (_Float16)(valid ? rot1 : -1.0f);
  o[2] = (_Float16)(valid ? t0 : -1.0f);
  o[3] = (_Float16)(valid ? t1 : -1.0f);
  o[4] = (_Float16)mask[(size_t)b * NPIX + n];
  #pragma unroll
  for (int c = 5; c < 16; ++c) o[c] = (_Float16)0.0f;
}

// pred_T[:, :-1] -> [B,1024,1,2]
__global__ void write_T_kernel(const float* __restrict__ pred, float* __restrict__ outT) {
  int i = blockIdx.x * blockDim.x + threadIdx.x;
  if (i >= BATCH * NUM_SEG) return;
  int b = i / NUM_SEG, s = i - b * NUM_SEG;
  outT[(size_t)i * 2 + 0] = pred[((size_t)b * 2 + 0) * S1 + s];
  outT[(size_t)i * 2 + 1] = pred[((size_t)b * 2 + 1) * S1 + s];
}

// ---------------------------------------------------------------------------
// Host side
// ---------------------------------------------------------------------------
struct UNetW {
  const float *e0w,*e0b,*e1w,*e1b,*e2w,*e2b,*e3w,*e3b,*e4w,*e4b;
  const float *d0w,*d0b,*d1w,*d1b,*d2w,*d2b,*d3w,*d3b,*ow,*ob;
};

// Dict params flattened in JAX pytree (sorted-key) order:
// d0b,d0w,d1b,d1w,d2b,d2w,d3b,d3w,e0b,e0w,e1b,e1w,e2b,e2w,e3b,e3w,e4b,e4w,ob,ow
static UNetW unpack_unet(void* const* din, int base) {
  UNetW u;
  u.d0b=(const float*)din[base+0];  u.d0w=(const float*)din[base+1];
  u.d1b=(const float*)din[base+2];  u.d1w=(const float*)din[base+3];
  u.d2b=(const float*)din[base+4];  u.d2w=(const float*)din[base+5];
  u.d3b=(const float*)din[base+6];  u.d3w=(const float*)din[base+7];
  u.e0b=(const float*)din[base+8];  u.e0w=(const float*)din[base+9];
  u.e1b=(const float*)din[base+10]; u.e1w=(const float*)din[base+11];
  u.e2b=(const float*)din[base+12]; u.e2w=(const float*)din[base+13];
  u.e3b=(const float*)din[base+14]; u.e3w=(const float*)din[base+15];
  u.e4b=(const float*)din[base+16]; u.e4w=(const float*)din[base+17];
  u.ob =(const float*)din[base+18]; u.ow =(const float*)din[base+19];
  return u;
}

struct Packs   { _Float16 *e0,*e1,*e2,*e3,*e4,*d3,*d2,*d1,*d0,*o; };
struct ActBufs { _Float16 *e0,*e1,*e2,*e3,*e4,*d3,*d2,*d1,*d0; };

template <int HOUT, int WOUT, int CH, int CinA, int CinB, int CinPad,
          int KF, int STRIDE, int UPA, bool GATED, int MBLK>
static void gl(const _Float16* a, const _Float16* b, const _Float16* wpk,
               const float* bias, _Float16* out, hipStream_t st) {
  static_assert((CH / 16) % MBLK == 0, "MBLK must divide CH/16");
  dim3 grid((unsigned)(BATCH * HOUT * WOUT / 16), (unsigned)(CH / (16 * MBLK)));
  gconv_wmma<HOUT, WOUT, CH, CinA, CinB, CinPad, KF, STRIDE, UPA, GATED, MBLK>
      <<<grid, 32, 0, st>>>(a, b, wpk, bias, out);
}

static void run_unet(const _Float16* in, const Packs& P, const UNetW& u,
                     const ActBufs& A, _Float16* feat, hipStream_t st) {
  gl<256,256, 16,  16,  0, 16, 3,1,0,true ,1>(in,   nullptr, P.e0, u.e0b, A.e0, st);
  gl<128,128, 32,  16,  0, 16, 3,2,0,true ,2>(A.e0, nullptr, P.e1, u.e1b, A.e1, st);
  gl< 64, 64, 64,  32,  0, 32, 3,2,0,true ,4>(A.e1, nullptr, P.e2, u.e2b, A.e2, st);
  gl< 32, 32,128,  64,  0, 64, 3,2,0,true ,4>(A.e2, nullptr, P.e3, u.e3b, A.e3, st);
  gl< 16, 16,256, 128,  0,128, 3,2,0,true ,4>(A.e3, nullptr, P.e4, u.e4b, A.e4, st);
  gl< 32, 32,128, 256,128,384, 3,1,1,true ,4>(A.e4, A.e3,    P.d3, u.d3b, A.d3, st);
  gl< 64, 64, 64, 128, 64,192, 3,1,1,true ,4>(A.d3, A.e2,    P.d2, u.d2b, A.d2, st);
  gl<128,128, 32,  64, 32, 96, 3,1,1,true ,2>(A.d2, A.e1,    P.d1, u.d1b, A.d1, st);
  gl<256,256, 16,  32, 16, 48, 3,1,1,true ,1>(A.d1, A.e0,    P.d0, u.d0b, A.d0, st);
  gl<256,256, 16,  16,  0, 16, 1,1,0,false,1>(A.d0, nullptr, P.o,  u.ob,  feat, st);
}

extern "C" void kernel_launch(void* const* d_in, const int* in_sizes, int n_in,
                              void* d_out, int out_size, void* d_ws, size_t ws_size,
                              hipStream_t stream) {
  (void)in_sizes; (void)n_in; (void)out_size; (void)ws_size;
  const float* src_pos = (const float*)d_in[0];
  const float* tar_pos = (const float*)d_in[1];
  const float* mask    = (const float*)d_in[2];
  const int*   overseg = (const int*)d_in[3];
  UNetW unet_r = unpack_unet(d_in, 4);
  const float* mr_b1 = (const float*)d_in[24];
  const float* mr_b2 = (const float*)d_in[25];
  const float* mr_w1 = (const float*)d_in[26];
  const float* mr_w2 = (const float*)d_in[27];
  UNetW unet_t = unpack_unet(d_in, 28);
  const float* mt_b1 = (const float*)d_in[48];
  const float* mt_b2 = (const float*)d_in[49];
  const float* mt_w1 = (const float*)d_in[50];
  const float* mt_w2 = (const float*)d_in[51];

  // ---- bump allocator over workspace ----
  char* wsp = (char*)d_ws;
  size_t off = 0;
  auto alloc = [&](size_t nbytes) -> void* {
    void* p = wsp + off;
    off += (nbytes + 255) & ~(size_t)255;
    return p;
  };

  int*   seg    = (int*)  alloc((size_t)BATCH * NPIX * sizeof(int));
  float* counts = (float*)alloc((size_t)BATCH * S1 * sizeof(float));
  float* meanS  = (float*)alloc((size_t)BATCH * 2 * S1 * sizeof(float));
  float* meanT  = (float*)alloc((size_t)BATCH * 2 * S1 * sizeof(float));
  _Float16* inbuf = (_Float16*)alloc((size_t)BATCH * NPIX * 16 * sizeof(_Float16));
  ActBufs A;
  A.e0 = (_Float16*)alloc((size_t)BATCH * NPIX       * 16  * sizeof(_Float16));
  A.e1 = (_Float16*)alloc((size_t)BATCH * (NPIX/4)   * 32  * sizeof(_Float16));
  A.e2 = (_Float16*)alloc((size_t)BATCH * (NPIX/16)  * 64  * sizeof(_Float16));
  A.e3 = (_Float16*)alloc((size_t)BATCH * (NPIX/64)  * 128 * sizeof(_Float16));
  A.e4 = (_Float16*)alloc((size_t)BATCH * (NPIX/256) * 256 * sizeof(_Float16));
  A.d3 = (_Float16*)alloc((size_t)BATCH * (NPIX/64)  * 128 * sizeof(_Float16));
  A.d2 = (_Float16*)alloc((size_t)BATCH * (NPIX/16)  * 64  * sizeof(_Float16));
  A.d1 = (_Float16*)alloc((size_t)BATCH * (NPIX/4)   * 32  * sizeof(_Float16));
  A.d0 = (_Float16*)alloc((size_t)BATCH * NPIX       * 16  * sizeof(_Float16));
  _Float16* feat = (_Float16*)alloc((size_t)BATCH * NPIX * 16 * sizeof(_Float16));
  float* featm = (float*)alloc((size_t)BATCH * 16 * S1 * sizeof(float));
  float* predR = (float*)alloc((size_t)BATCH * 2 * S1 * sizeof(float));
  float* predT = (float*)alloc((size_t)BATCH * 2 * S1 * sizeof(float));

  const int TPB = 256;
  const int totalPix = BATCH * NPIX;
  const int gPix = (totalPix + TPB - 1) / TPB;

  // ---- repack all conv weights into WMMA A-fragment order (f16) ----
  auto do_pack = [&](const float* w, int coutTot, int cinT, int cinPad, int kk) -> _Float16* {
    int nch = (cinPad * kk + 31) / 32;
    size_t halves = (size_t)nch * (coutTot >> 4) * 512;
    _Float16* p = (_Float16*)alloc(halves * sizeof(_Float16));
    pack_w_kernel<<<((int)halves + TPB - 1) / TPB, TPB, 0, stream>>>(
        w, p, coutTot, cinT, cinPad, kk, nch);
    return p;
  };
  auto pack_unet = [&](const UNetW& u) -> Packs {
    Packs P;
    P.e0 = do_pack(u.e0w,  32,   5,  16, 9);
    P.e1 = do_pack(u.e1w,  64,  16,  16, 9);
    P.e2 = do_pack(u.e2w, 128,  32,  32, 9);
    P.e3 = do_pack(u.e3w, 256,  64,  64, 9);
    P.e4 = do_pack(u.e4w, 512, 128, 128, 9);
    P.d3 = do_pack(u.d3w, 256, 384, 384, 9);
    P.d2 = do_pack(u.d2w, 128, 192, 192, 9);
    P.d1 = do_pack(u.d1w,  64,  96,  96, 9);
    P.d0 = do_pack(u.d0w,  32,  48,  48, 9);
    P.o  = do_pack(u.ow,   16,  16,  16, 1);
    return P;
  };
  Packs PR = pack_unet(unet_r);
  Packs PT = pack_unet(unet_t);

  // 1) seg fixup (invalid pixels -> fake bucket NUM_SEG)
  prep_seg_kernel<<<gPix, TPB, 0, stream>>>(tar_pos, overseg, seg, totalPix);

  // 2) centroids: scatter-mean of src_pos and tar_pos over segments
  zero_kernel<<<(BATCH*S1 + TPB-1)/TPB, TPB, 0, stream>>>(counts, BATCH*S1);
  zero_kernel<<<(BATCH*2*S1 + TPB-1)/TPB, TPB, 0, stream>>>(meanS, BATCH*2*S1);
  zero_kernel<<<(BATCH*2*S1 + TPB-1)/TPB, TPB, 0, stream>>>(meanT, BATCH*2*S1);
  seg_count_kernel<<<gPix, TPB, 0, stream>>>(seg, counts, totalPix);
  seg_sum_kernel<<<gPix, TPB, 0, stream>>>(src_pos, seg, meanS, 2, totalPix);
  seg_sum_kernel<<<gPix, TPB, 0, stream>>>(tar_pos, seg, meanT, 2, totalPix);
  seg_mean_kernel<<<(BATCH*2*S1 + TPB-1)/TPB, TPB, 0, stream>>>(meanS, counts, 2);
  seg_mean_kernel<<<(BATCH*2*S1 + TPB-1)/TPB, TPB, 0, stream>>>(meanT, counts, 2);

  // 3) UNet-R input and forward
  build_in0_kernel<<<gPix, TPB, 0, stream>>>(src_pos, tar_pos, mask, seg,
                                             meanS, meanT, inbuf, totalPix);
  run_unet(inbuf, PR, unet_r, A, feat, stream);

  // 4) per-segment features -> rotation MLP
  zero_kernel<<<(BATCH*16*S1 + TPB-1)/TPB, TPB, 0, stream>>>(featm, BATCH*16*S1);
  seg_sum_feat_kernel<<<gPix, TPB, 0, stream>>>(feat, seg, featm, totalPix);
  seg_mean_kernel<<<(BATCH*16*S1 + TPB-1)/TPB, TPB, 0, stream>>>(featm, counts, 16);
  mreg_kernel<<<(BATCH*S1 + TPB-1)/TPB, TPB, 0, stream>>>(featm, mr_w1, mr_b1,
                                                          mr_w2, mr_b2, predR);
  float* outR = (float*)d_out;                       // [B,1024,2,2]
  float* outT = outR + (size_t)BATCH * NUM_SEG * 4;  // [B,1024,1,2]
  write_R_kernel<<<(BATCH*NUM_SEG + TPB-1)/TPB, TPB, 0, stream>>>(predR, outR);

  // 5) rotate, UNet-T input and forward
  build_pos1_kernel<<<gPix, TPB, 0, stream>>>(src_pos, tar_pos, mask, seg,
                                              predR, inbuf, totalPix);
  run_unet(inbuf, PT, unet_t, A, feat, stream);

  // 6) per-segment features -> translation MLP
  zero_kernel<<<(BATCH*16*S1 + TPB-1)/TPB, TPB, 0, stream>>>(featm, BATCH*16*S1);
  seg_sum_feat_kernel<<<gPix, TPB, 0, stream>>>(feat, seg, featm, totalPix);
  seg_mean_kernel<<<(BATCH*16*S1 + TPB-1)/TPB, TPB, 0, stream>>>(featm, counts, 16);
  mreg_kernel<<<(BATCH*S1 + TPB-1)/TPB, TPB, 0, stream>>>(featm, mt_w1, mt_b1,
                                                          mt_w2, mt_b2, predT);
  write_T_kernel<<<(BATCH*NUM_SEG + TPB-1)/TPB, TPB, 0, stream>>>(predT, outT);
}
